// CausalSelfAttention_26422638805370
// MI455X (gfx1250) — compile-verified, matrix operand order checked by reading, not run on hardware
//
#include <hip/hip_runtime.h>
#include <hip/hip_bf16.h>

// ---------------------------------------------------------------------------
// CausalSelfAttention forward, MI455X (gfx1250, wave32, WMMA bf16->f32)
//   B=2, T=2048, C=2048, H=16, hd=128
// Pipeline: cvt(bf16) -> QKV WMMA-GEMM (TDM double-buffered LDS staging,
//           scatter epilogue, V transposed) -> RoPE in-place ->
//           flash attention (TDM-staged K/V tiles, WMMA) -> out-proj GEMM
// ---------------------------------------------------------------------------

typedef __bf16 bf16_t;
typedef __attribute__((ext_vector_type(8)))  bf16_t v8bf;
typedef __attribute__((ext_vector_type(16))) bf16_t v16bf;
typedef __attribute__((ext_vector_type(8)))  float  v8f;
typedef __attribute__((ext_vector_type(4)))  unsigned int v4u;
typedef __attribute__((ext_vector_type(8)))  int v8i;
typedef __attribute__((ext_vector_type(4)))  int v4i;

#define T_SEQ 2048
#define HD    128
#define NH    16
#define CDIM  2048

#if defined(__has_builtin)
#if __has_builtin(__builtin_amdgcn_tensor_load_to_lds) && \
    __has_builtin(__builtin_amdgcn_s_wait_tensorcnt)
#define USE_TDM 1
#endif
#endif

// ---------------------------------------------------------------------------
// Issue a 2D TDM tile load: rows x cols bf16 elements, global row stride
// `gstride` elements, packed contiguously into LDS (row-major, ld = cols).
// D# packing per CDNA5 ISA (2D: groups 2/3 zero).
// This toolchain exposes the 6-arg builtin:
//   (uint32x4 g0, int32x8 g1, int32x4, int32x4, int32x8, i32 cpol)
#ifdef USE_TDM
__device__ inline void tdm_load_2d(unsigned lds_off, const bf16_t* gptr,
                                   unsigned rows, unsigned cols,
                                   unsigned gstride_elems) {
  unsigned long long ga = (unsigned long long)(uintptr_t)gptr;
  v4u g0;
  g0[0] = 1u;                                   // count=1, user descriptor
  g0[1] = lds_off;                              // lds_addr (bytes)
  g0[2] = (unsigned)(ga & 0xffffffffu);         // global_addr[31:0]
  g0[3] = (unsigned)((ga >> 32) & 0x01ffffffu)  // global_addr[56:32]
        | (2u << 30);                           // type = 2 ("image")
  v8i g1;
  g1[0] = (int)(1u << 16);                      // data_size=1 (2 bytes/elem)
  g1[1] = (int)(cols << 16);                    // tensor_dim0[15:0] @ bits63:48
  g1[2] = (int)(rows << 16);                    // tensor_dim1[15:0] @ bits95:80
  g1[3] = (int)(cols << 16);                    // tile_dim0 @ bits127:112
  g1[4] = (int)rows;                            // tile_dim1 @ bits143:128
  g1[5] = (int)gstride_elems;                   // tensor_dim0_stride[31:0]
  g1[6] = 0;
  g1[7] = 0;
  v4i gz4 = {0, 0, 0, 0};                       // 2D: groups 2/3 unused
  v8i gz8 = {0, 0, 0, 0, 0, 0, 0, 0};
  __builtin_amdgcn_tensor_load_to_lds(g0, g1, gz4, gz4, gz8, 0);
}
__device__ inline unsigned lds_offset_of(const void* p) {
  return (unsigned)(uintptr_t)p;                // LDS aperture: addr[31:0]
}
#endif

// ---------------------------------------------------------------------------
// 16x32 bf16 fragment load (A-pattern; identical for B of D=A*B^T with
// row-major storage). lane&15 = row, lane>>4 = K-group.
__device__ inline v16bf load_frag(const bf16_t* __restrict__ base, int ld) {
  const int lane = threadIdx.x & 31;
  const int row  = lane & 15;
  const int g    = lane >> 4;
  const bf16_t* p = base + (size_t)row * ld + g * 8;
  v8bf lo = *(const v8bf*)(p);
  v8bf hi = *(const v8bf*)(p + 16);
  return __builtin_shufflevector(lo, hi, 0,1,2,3,4,5,6,7,8,9,10,11,12,13,14,15);
}

__device__ inline v8f wmma_bf16(v16bf a, v16bf b, v8f c) {
  return __builtin_amdgcn_wmma_f32_16x16x32_bf16(false, a, false, b, (short)0, c,
                                                 false, false);
}

// ---------------------------------------------------------------------------
__global__ void cvt_f32_bf16(const float* __restrict__ src,
                             bf16_t* __restrict__ dst, int n) {
  int i = blockIdx.x * blockDim.x + threadIdx.x;
  if (i < n) dst[i] = (bf16_t)src[i];
}

// ---------------------------------------------------------------------------
// WMMA GEMM: out[m,n] = sum_k A[m,k]*B[n,k].  Block = 8 waves -> 128x128 tile.
// A/B K-slabs (128x32 bf16 each) staged to LDS by TDM, double-buffered.
// MODE 0: f32 store.  MODE 1: QKV scatter (v stored transposed d-major).
template <int MODE>
__global__ __launch_bounds__(256) void gemm_bf16(
    const bf16_t* __restrict__ A, const bf16_t* __restrict__ B,
    float* __restrict__ out_f32,
    bf16_t* __restrict__ qb, bf16_t* __restrict__ kb, bf16_t* __restrict__ vtb,
    int N, int K) {
  __shared__ bf16_t smA[2][128 * 32];
  __shared__ bf16_t smB[2][128 * 32];

  const int wave  = threadIdx.x >> 5;
  const int lane  = threadIdx.x & 31;
  const int mrow0 = blockIdx.y * 128;
  const int nbase = blockIdx.x * 128;
  const int mbase = mrow0 + wave * 16;

  v8f acc[8];
#pragma unroll
  for (int nt = 0; nt < 8; ++nt) acc[nt] = {0.f,0.f,0.f,0.f,0.f,0.f,0.f,0.f};

#ifdef USE_TDM
  const int S = K / 32;
  if (wave == 0) {
    tdm_load_2d(lds_offset_of(&smA[0][0]), A + (size_t)mrow0 * K, 128, 32, K);
    tdm_load_2d(lds_offset_of(&smB[0][0]), B + (size_t)nbase * K, 128, 32, K);
  }
  for (int s = 0; s < S; ++s) {
    if (wave == 0) {
      if (s + 1 < S) {
        int nb = (s + 1) & 1;
        tdm_load_2d(lds_offset_of(&smA[nb][0]),
                    A + (size_t)mrow0 * K + (s + 1) * 32, 128, 32, K);
        tdm_load_2d(lds_offset_of(&smB[nb][0]),
                    B + (size_t)nbase * K + (s + 1) * 32, 128, 32, K);
        __builtin_amdgcn_s_wait_tensorcnt((short)2);
      } else {
        __builtin_amdgcn_s_wait_tensorcnt((short)0);
      }
    }
    __syncthreads();
    const bf16_t* As = &smA[s & 1][0];
    const bf16_t* Bs = &smB[s & 1][0];
    v16bf af = load_frag(As + wave * 16 * 32, 32);
#pragma unroll
    for (int nt = 0; nt < 8; ++nt) {
      v16bf bfr = load_frag(Bs + nt * 16 * 32, 32);
      acc[nt] = wmma_bf16(af, bfr, acc[nt]);
    }
    __syncthreads();
  }
#else
  for (int kk = 0; kk < K; kk += 32) {
    v16bf af = load_frag(A + (size_t)mbase * K + kk, K);
    __builtin_prefetch(A + (size_t)mbase * K + kk + 128, 0, 1);
#pragma unroll
    for (int nt = 0; nt < 8; ++nt) {
      v16bf bfr = load_frag(B + (size_t)(nbase + nt * 16) * K + kk, K);
      acc[nt] = wmma_bf16(af, bfr, acc[nt]);
    }
  }
#endif

  const int rbase = 8 * (lane >> 4);  // C layout: rows r+rbase, col = lane&15
  const int nl    = lane & 15;
#pragma unroll
  for (int nt = 0; nt < 8; ++nt) {
#pragma unroll
    for (int r = 0; r < 8; ++r) {
      float v = acc[nt][r];
      int m = mbase + r + rbase;
      int n = nbase + nt * 16 + nl;
      if (MODE == 0) {
        out_f32[(size_t)m * N + n] = v;
      } else {
        int which = n >> 11;            // 0=q 1=k 2=v
        int rem   = n & 2047;
        int h     = rem >> 7;
        int dd    = rem & 127;
        int b     = m >> 11;
        int t     = m & 2047;
        if (which == 2) {
          vtb[((size_t)((b * NH + h) * HD + dd)) * T_SEQ + t] = (bf16_t)v;
        } else {
          size_t idx = ((size_t)((b * NH + h) * T_SEQ + t)) * HD + dd;
          if (which == 0) qb[idx] = (bf16_t)v;
          else            kb[idx] = (bf16_t)v;
        }
      }
    }
  }
}

// ---------------------------------------------------------------------------
// Interleaved RoPE, in-place on bf16 q/k in (B,H,T,hd) layout.
__global__ void rope_kernel(bf16_t* __restrict__ q, bf16_t* __restrict__ k,
                            const float* __restrict__ cosv,
                            const float* __restrict__ sinv) {
  int i  = blockIdx.x * blockDim.x + threadIdx.x;  // over 2*16*2048*64
  int j  = i & 63;
  int t  = (i >> 6) & 2047;
  int bh = i >> 17;  // 0..31
  float c = cosv[t * 64 + j];
  float s = sinv[t * 64 + j];
  size_t base = ((size_t)bh * T_SEQ + t) * HD + 2 * j;
  float q1 = (float)q[base], q2 = (float)q[base + 1];
  q[base]     = (bf16_t)(q1 * c - q2 * s);
  q[base + 1] = (bf16_t)(q1 * s + q2 * c);
  float k1 = (float)k[base], k2 = (float)k[base + 1];
  k[base]     = (bf16_t)(k1 * c - k2 * s);
  k[base + 1] = (bf16_t)(k1 * s + k2 * c);
}

// ---------------------------------------------------------------------------
// Flash attention. Grid (T/128, H, B); block = 256 = 8 waves.
// Wave owns 16 q-rows x hd=128 in registers; K/V^T 32-wide k-slabs staged to
// LDS by TDM (shared by all 8 waves), double-buffered. Online softmax fp32.
__global__ __launch_bounds__(256) void flash_attn(
    const bf16_t* __restrict__ q, const bf16_t* __restrict__ k,
    const bf16_t* __restrict__ vt, bf16_t* __restrict__ out) {
  __shared__ bf16_t smK[2][32 * 128];   // rows = k-pos (32), ld = 128
  __shared__ bf16_t smV[2][128 * 32];   // rows = d (128),   ld = 32
  __shared__ bf16_t lds_p[8][16 * 32];  // per-wave P staging

  const int wave = threadIdx.x >> 5;
  const int lane = threadIdx.x & 31;
  const int bq   = blockIdx.x;
  const int h    = blockIdx.y;
  const int b    = blockIdx.z;
  const int qbase = bq * 128 + wave * 16;

  const bf16_t* qp = q  + ((size_t)(b * NH + h) * T_SEQ) * HD;
  const bf16_t* kp = k  + ((size_t)(b * NH + h) * T_SEQ) * HD;
  const bf16_t* vp = vt + ((size_t)(b * NH + h) * HD) * T_SEQ;

  v16bf qa[4];
#pragma unroll
  for (int j = 0; j < 4; ++j)
    qa[j] = load_frag(qp + (size_t)qbase * HD + j * 32, HD);

  v8f o[8];
#pragma unroll
  for (int nt = 0; nt < 8; ++nt) o[nt] = {0.f,0.f,0.f,0.f,0.f,0.f,0.f,0.f};

  float mrow[8], lrow[8];
#pragma unroll
  for (int r = 0; r < 8; ++r) { mrow[r] = -1e30f; lrow[r] = 0.f; }

  const int rbase = 8 * (lane >> 4);
  const int nl    = lane & 15;
  const float scale = 0.08838834764831845f;  // 1/sqrt(128)

  const int S = (bq + 1) * 4;  // number of 32-wide k-slabs (block-uniform)

#ifdef USE_TDM
  if (wave == 0) {
    tdm_load_2d(lds_offset_of(&smK[0][0]), kp, 32, 128, HD);
    tdm_load_2d(lds_offset_of(&smV[0][0]), vp, 128, 32, T_SEQ);
  }
#endif

  for (int s = 0; s < S; ++s) {
    const int kk = s * 32;
#ifdef USE_TDM
    if (wave == 0) {
      if (s + 1 < S) {
        int nb = (s + 1) & 1;
        tdm_load_2d(lds_offset_of(&smK[nb][0]), kp + (size_t)(kk + 32) * HD,
                    32, 128, HD);
        tdm_load_2d(lds_offset_of(&smV[nb][0]), vp + (kk + 32),
                    128, 32, T_SEQ);
        __builtin_amdgcn_s_wait_tensorcnt((short)2);
      } else {
        __builtin_amdgcn_s_wait_tensorcnt((short)0);
      }
    }
    __syncthreads();
    const bf16_t* Ks = &smK[s & 1][0];
    const bf16_t* Vs = &smV[s & 1][0];
#endif

    // S = Q * K^T for 32 k-columns (two 16-wide tiles)
    v8f s0 = {0.f,0.f,0.f,0.f,0.f,0.f,0.f,0.f};
    v8f s1 = {0.f,0.f,0.f,0.f,0.f,0.f,0.f,0.f};
#pragma unroll
    for (int j = 0; j < 4; ++j) {
#ifdef USE_TDM
      v16bf kb0 = load_frag(Ks + j * 32, 128);
      v16bf kb1 = load_frag(Ks + 16 * 128 + j * 32, 128);
#else
      v16bf kb0 = load_frag(kp + (size_t)kk * HD + j * 32, HD);
      v16bf kb1 = load_frag(kp + (size_t)(kk + 16) * HD + j * 32, HD);
#endif
      s0 = wmma_bf16(qa[j], kb0, s0);
      s1 = wmma_bf16(qa[j], kb1, s1);
    }

    // scale + causal mask + online softmax (rows live in 16-lane halves)
    float p0[8], p1[8], fac[8];
#pragma unroll
    for (int r = 0; r < 8; ++r) {
      int qrow = qbase + r + rbase;
      int c0 = kk + nl;
      int c1 = kk + 16 + nl;
      float a0 = s0[r] * scale; if (c0 > qrow) a0 = -1e30f;
      float a1 = s1[r] * scale; if (c1 > qrow) a1 = -1e30f;
      float mx = fmaxf(a0, a1);
#pragma unroll
      for (int off = 1; off < 16; off <<= 1)
        mx = fmaxf(mx, __shfl_xor(mx, off, 32));
      float mnew = fmaxf(mrow[r], mx);
      float f  = __expf(mrow[r] - mnew);
      float e0 = __expf(a0 - mnew);
      float e1 = __expf(a1 - mnew);
      float sum = e0 + e1;
#pragma unroll
      for (int off = 1; off < 16; off <<= 1)
        sum += __shfl_xor(sum, off, 32);
      lrow[r] = lrow[r] * f + sum;
      mrow[r] = mnew;
      fac[r] = f; p0[r] = e0; p1[r] = e1;
    }

#pragma unroll
    for (int nt = 0; nt < 8; ++nt)
#pragma unroll
      for (int r = 0; r < 8; ++r) o[nt][r] *= fac[r];

    // stage P (C-layout) -> LDS -> A-fragment layout (per-wave region)
    bf16_t* lp = &lds_p[wave][0];
#pragma unroll
    for (int r = 0; r < 8; ++r) {
      int m = r + rbase;
      lp[m * 32 + nl]      = (bf16_t)p0[r];
      lp[m * 32 + 16 + nl] = (bf16_t)p1[r];
    }
    v16bf pa = load_frag(lp, 32);

    // O += P * V  (V^T row-major: row = d, cols = t)
#pragma unroll
    for (int nt = 0; nt < 8; ++nt) {
#ifdef USE_TDM
      v16bf vb = load_frag(Vs + nt * 16 * 32, 32);
#else
      v16bf vb = load_frag(vp + (size_t)(nt * 16) * T_SEQ + kk, T_SEQ);
#endif
      o[nt] = wmma_bf16(pa, vb, o[nt]);
    }
#ifdef USE_TDM
    __syncthreads();
#endif
  }

  // normalize + store to (B,T,C) bf16 for the out-projection GEMM
#pragma unroll
  for (int nt = 0; nt < 8; ++nt) {
#pragma unroll
    for (int r = 0; r < 8; ++r) {
      int m = qbase + r + rbase;
      int d = h * HD + nt * 16 + nl;
      out[((size_t)(b * T_SEQ + m)) * CDIM + d] = (bf16_t)(o[nt][r] / lrow[r]);
    }
  }
}

// ---------------------------------------------------------------------------
extern "C" void kernel_launch(void* const* d_in, const int* in_sizes, int n_in,
                              void* d_out, int out_size, void* d_ws, size_t ws_size,
                              hipStream_t stream) {
  const float* x     = (const float*)d_in[0];  // (2,2048,2048)
  const float* cosv  = (const float*)d_in[1];  // (2048,64)
  const float* sinv  = (const float*)d_in[2];  // (2048,64)
  const float* wqkv  = (const float*)d_in[3];  // (6144,2048)
  const float* wproj = (const float*)d_in[4];  // (2048,2048)
  float* out = (float*)d_out;                  // (2,2048,2048)

  const size_t N_X    = (size_t)2 * T_SEQ * CDIM;       // 8,388,608
  const size_t N_WQKV = (size_t)3 * CDIM * CDIM;        // 12,582,912
  const size_t N_WPRJ = (size_t)CDIM * CDIM;            // 4,194,304
  const size_t N_QKV1 = (size_t)2 * NH * T_SEQ * HD;    // 8,388,608

  bf16_t* p = (bf16_t*)d_ws;
  bf16_t* xb    = p; p += N_X;
  bf16_t* wqkvb = p; p += N_WQKV;
  bf16_t* wprjb = p; p += N_WPRJ;
  bf16_t* qb    = p; p += N_QKV1;
  bf16_t* kb    = p; p += N_QKV1;
  bf16_t* vtb   = p; p += N_QKV1;
  bf16_t* aob   = p; p += N_X;

  // 1) fp32 -> bf16
  cvt_f32_bf16<<<(int)((N_X    + 255) / 256), 256, 0, stream>>>(x,     xb,    (int)N_X);
  cvt_f32_bf16<<<(int)((N_WQKV + 255) / 256), 256, 0, stream>>>(wqkv,  wqkvb, (int)N_WQKV);
  cvt_f32_bf16<<<(int)((N_WPRJ + 255) / 256), 256, 0, stream>>>(wproj, wprjb, (int)N_WPRJ);

  // 2) QKV projection (M=4096, N=6144, K=2048), scatter epilogue
  gemm_bf16<1><<<dim3(48, 32), 256, 0, stream>>>(xb, wqkvb, nullptr,
                                                 qb, kb, vtb, 6144, 2048);

  // 3) RoPE in-place on q,k
  rope_kernel<<<16384, 256, 0, stream>>>(qb, kb, cosv, sinv);

  // 4) flash attention -> (B,T,C) bf16
  flash_attn<<<dim3(16, NH, 2), 256, 0, stream>>>(qb, kb, vtb, aob);

  // 5) output projection (M=4096, N=2048, K=2048) -> f32
  gemm_bf16<0><<<dim3(16, 32), 256, 0, stream>>>(aob, wprjb, out,
                                                 nullptr, nullptr, nullptr,
                                                 2048, 2048);
}
